// MultiHeadAttention_10428180594805
// MI455X (gfx1250) — matrix operand order repaired and checked by reading, not compiled, not run on hardware
//
#include <hip/hip_runtime.h>
#include <math.h>

// ---------------------------------------------------------------------------
// MI455X (gfx1250) multi-head attention, wave32 + V_WMMA_F32_16X16X32_F16.
// B=2, S=2048, D=1024, H=16, DK=64. f32 in/out, f16 compute / f32 accumulate.
// Round 3: GEMM K-loop fully unrolled with parity double-buffers (no
// loop-carried fragment swaps -> no acc copies / hazard NOPs, immediate-offset
// addressing for all 192 loads).
// ---------------------------------------------------------------------------

typedef __attribute__((ext_vector_type(16))) _Float16 v16h;
typedef __attribute__((ext_vector_type(8)))  _Float16 v8h;
typedef __attribute__((ext_vector_type(8)))  float    v8f;

#define B_    2
#define S_    2048
#define D_    1024
#define H_    16
#define DK_   64
#define MTOT  (B_ * S_)          // 4096 rows of X

union FragU { v16h v; v8h h8[2]; };

// A fragment: 16(M) x 32(K) f16 tile, row-major source, leading dim `ld`
// (halves), tile origin p. Lane l: row = l&15, contiguous
// K run 16*(l>>4) .. +15 -> elems 0..15.
__device__ __forceinline__ v16h load_frag_a(const _Float16* __restrict__ p,
                                            int ld, int lane) {
    const int row = lane & 15, kg = lane >> 4;
    FragU u;
    u.h8[0] = *(const v8h*)(p + (size_t)row * ld + kg * 16);
    u.h8[1] = *(const v8h*)(p + (size_t)row * ld + kg * 16 + 8);
    return u.v;
}

// B fragment: 32(K) x 16(N) where B[k][n] = src[n][k] (src row-major, ld in
// halves, tile origin p = &src[n0][k0]). Lane l: column n = l&15, contiguous
// K run 16*(l>>4) .. +15.
__device__ __forceinline__ v16h load_frag_b(const _Float16* __restrict__ p,
                                            int ld, int lane) {
    const int n = lane & 15, kg = lane >> 4;
    const _Float16* q = p + (size_t)n * ld + kg * 16;
    FragU u;
    u.h8[0] = *(const v8h*)(q);
    u.h8[1] = *(const v8h*)(q + 8);
    return u.v;
}

__device__ __forceinline__ v8f wmma_f16(v16h a, v16h b, v8f c) {
    return __builtin_amdgcn_wmma_f32_16x16x32_f16(
        /*neg_a=*/false, a, /*neg_b=*/false, b,
        /*c_mod=*/(short)0, c, /*reuse_a=*/false, /*reuse_b=*/false);
}

// ---------------------------------------------------------------------------
// Kernel 0: f32 -> f16 convert (X and the four weight matrices).
// ---------------------------------------------------------------------------
__global__ void __launch_bounds__(256)
cvt_f32_to_f16(const float* __restrict__ in, _Float16* __restrict__ out, int n) {
    int i = blockIdx.x * blockDim.x + threadIdx.x;
    int stride = gridDim.x * blockDim.x;
    for (; i < n; i += stride) out[i] = (_Float16)in[i];
}

// ---------------------------------------------------------------------------
// Shared GEMM core: per-wave 32(M) x 64(N) tile, K = D_ in 32-steps.
// Fully unrolled with parity-indexed double buffers: every fragment is an SSA
// value (no loop-carried swaps), every load is base + immediate offset, and
// each step's 8 WMMAs overlap the next step's 6 fragment loads.
// ---------------------------------------------------------------------------
__device__ __forceinline__ void gemm_core_32x64(
    const _Float16* __restrict__ Arow,   // &A[m0][0], ld = D_
    const _Float16* __restrict__ Brow,   // &W[n0][0], ld = D_
    int lane, v8f (&acc)[2][4]) {
    v16h a[2][2], b[2][4];
    a[0][0] = load_frag_a(Arow,           D_, lane);
    a[0][1] = load_frag_a(Arow + 16 * D_, D_, lane);
    b[0][0] = load_frag_b(Brow,           D_, lane);
    b[0][1] = load_frag_b(Brow + 16 * D_, D_, lane);
    b[0][2] = load_frag_b(Brow + 32 * D_, D_, lane);
    b[0][3] = load_frag_b(Brow + 48 * D_, D_, lane);

#pragma unroll
    for (int i = 0; i < D_ / 32; ++i) {
        const int cur = i & 1, nxt = cur ^ 1;
        if (i + 1 < D_ / 32) {
            const int k = 32 * (i + 1);
            a[nxt][0] = load_frag_a(Arow + k,           D_, lane);
            a[nxt][1] = load_frag_a(Arow + 16 * D_ + k, D_, lane);
            b[nxt][0] = load_frag_b(Brow + k,           D_, lane);
            b[nxt][1] = load_frag_b(Brow + 16 * D_ + k, D_, lane);
            b[nxt][2] = load_frag_b(Brow + 32 * D_ + k, D_, lane);
            b[nxt][3] = load_frag_b(Brow + 48 * D_ + k, D_, lane);
        }
#pragma unroll
        for (int r = 0; r < 2; ++r) {
#pragma unroll
            for (int c = 0; c < 4; ++c) {
                acc[r][c] = wmma_f16(a[cur][r], b[cur][c], acc[r][c]);
            }
        }
    }
}

// ---------------------------------------------------------------------------
// Kernel 1: fused QKV projection.  C[m,n] = sum_k X[m,k] * W[n,k] + b[n].
// 256 threads = 8 waves as 4(M) x 2(N); block covers 128(M) x 128(N).
// grid.z in {0,1,2} selects Q/K/V.
//   Q -> [B,H,S,DK] scaled by 1/sqrt(DK)
//   K -> [B,H,S,DK]
//   V -> [B,H,DK,S]   (transposed so PV B-fragments are contiguous)
// ---------------------------------------------------------------------------
__global__ void __launch_bounds__(256)
qkv_proj_kernel(const _Float16* __restrict__ Xh,
                const _Float16* __restrict__ W3,   // Wq,Wk,Wv consecutive
                const float* __restrict__ bq,
                const float* __restrict__ bk,
                const float* __restrict__ bv,
                _Float16* __restrict__ Qh,
                _Float16* __restrict__ Kh,
                _Float16* __restrict__ Vt) {
    const int lane = threadIdx.x & 31;
    const int wave = threadIdx.x >> 5;
    const int m0 = blockIdx.x * 128 + (wave & 3) * 32;
    const int n0 = blockIdx.y * 128 + (wave >> 2) * 64;
    const int z  = blockIdx.z;
    const _Float16* W = W3 + (size_t)z * D_ * D_;
    const float* bias = (z == 0) ? bq : (z == 1) ? bk : bv;

    v8f acc[2][4] = {};
    gemm_core_32x64(Xh + (size_t)m0 * D_, W + (size_t)n0 * D_, lane, acc);

    const int col = lane & 15, grp = lane >> 4;
#pragma unroll
    for (int r = 0; r < 2; ++r) {
#pragma unroll
        for (int c = 0; c < 4; ++c) {
            const int n = n0 + 16 * c + col;
            const int h = n >> 6;          // n / DK
            const int d = n & 63;          // n % DK
            const float bn = bias[n];
#pragma unroll
            for (int j = 0; j < 8; ++j) {
                const int m = m0 + 16 * r + j + 8 * grp;
                const int b = m >> 11;     // m / S
                const int s = m & 2047;    // m % S
                const float v = acc[r][c][j] + bn;
                if (z == 0) {
                    Qh[((size_t)((b * H_ + h) * S_ + s)) * DK_ + d] =
                        (_Float16)(v * 0.125f);   // 1/sqrt(64)
                } else if (z == 1) {
                    Kh[((size_t)((b * H_ + h) * S_ + s)) * DK_ + d] = (_Float16)v;
                } else {
                    Vt[((size_t)((b * H_ + h) * DK_ + d)) * S_ + s] = (_Float16)v;
                }
            }
        }
    }
}

// ---------------------------------------------------------------------------
// Attention KV-block body.  On entry kf[] holds K fragments for block kv0;
// on exit kf[] holds fragments for block kvn (prefetched early, so the loads
// overlap the softmax VALU work and the LDS round-trip).
// ---------------------------------------------------------------------------
template <bool MASKED>
__device__ __forceinline__ void attn_block(
    const _Float16* __restrict__ Kp, const _Float16* __restrict__ Vp,
    _Float16* pl, v16h aq0, v16h aq1, v16h (&kf)[4],
    int kv0, int kvn, int q0, int lane, int col, int grp,
    v8f (&o)[4], float (&mrun)[8], float (&lrun)[8]) {
    // ---- scores: S = Q(16x64) . K_blk^T(64x32), two 16-col chunks ----
    v8f s0 = {}, s1 = {};
    s0 = wmma_f16(aq0, kf[0], s0);
    s0 = wmma_f16(aq1, kf[1], s0);
    s1 = wmma_f16(aq0, kf[2], s1);
    s1 = wmma_f16(aq1, kf[3], s1);

    // ---- issue V loads for this block and K prefetch for the next block
    //      now, so they fly during the softmax VALU section ----
    v16h vf0 = load_frag_b(Vp + (size_t)0  * S_ + kv0, S_, lane);
    v16h vf1 = load_frag_b(Vp + (size_t)16 * S_ + kv0, S_, lane);
    v16h vf2 = load_frag_b(Vp + (size_t)32 * S_ + kv0, S_, lane);
    v16h vf3 = load_frag_b(Vp + (size_t)48 * S_ + kv0, S_, lane);
    kf[0] = load_frag_b(Kp + (size_t)(kvn)*DK_ + 0,       DK_, lane);
    kf[1] = load_frag_b(Kp + (size_t)(kvn)*DK_ + 32,      DK_, lane);
    kf[2] = load_frag_b(Kp + (size_t)(kvn + 16) * DK_ + 0,  DK_, lane);
    kf[3] = load_frag_b(Kp + (size_t)(kvn + 16) * DK_ + 32, DK_, lane);

    // ---- (mask +) online softmax update; C-layout row = j + 8*grp ----
#pragma unroll
    for (int j = 0; j < 8; ++j) {
        if (MASKED) {
            const int q = q0 + j + 8 * grp;
            if (kv0 + col > q)      s0[j] = -1e30f;
            if (kv0 + 16 + col > q) s1[j] = -1e30f;
        }
        float mx = fmaxf(s0[j], s1[j]);
        mx = fmaxf(mx, __shfl_xor(mx, 1, 32));
        mx = fmaxf(mx, __shfl_xor(mx, 2, 32));
        mx = fmaxf(mx, __shfl_xor(mx, 4, 32));
        mx = fmaxf(mx, __shfl_xor(mx, 8, 32));
        const float mnew = fmaxf(mrun[j], mx);

        const float p0 = __expf(s0[j] - mnew);
        const float p1 = __expf(s1[j] - mnew);
        float rs = p0 + p1;
        rs += __shfl_xor(rs, 1, 32);
        rs += __shfl_xor(rs, 2, 32);
        rs += __shfl_xor(rs, 4, 32);
        rs += __shfl_xor(rs, 8, 32);

        const float scale = __expf(mrun[j] - mnew);
        lrun[j] = lrun[j] * scale + rs;
        mrun[j] = mnew;
#pragma unroll
        for (int c = 0; c < 4; ++c) o[c][j] *= scale;

        // Stage P (f16) row-major [16][32] into this wave's LDS slice.
        pl[(j + 8 * grp) * 32 + col]      = (_Float16)p0;
        pl[(j + 8 * grp) * 32 + 16 + col] = (_Float16)p1;
    }

    // DS ops are in-order per wave; wait so the A-frag load sees all lanes.
    asm volatile("s_wait_dscnt 0" ::: "memory");

    // ---- O += P(16x32) . V_blk(32x64) ----
    const v16h ap = load_frag_a(pl, 32, lane);
    o[0] = wmma_f16(ap, vf0, o[0]);
    o[1] = wmma_f16(ap, vf1, o[1]);
    o[2] = wmma_f16(ap, vf2, o[2]);
    o[3] = wmma_f16(ap, vf3, o[3]);
    asm volatile("s_wait_dscnt 0" ::: "memory");  // frag consumed before reuse
}

// ---------------------------------------------------------------------------
// Kernel 2: causal flash attention. One wave per 16-row query tile; online
// softmax over 32-wide KV blocks. Causal split: unmasked main loop + exactly
// one masked diagonal block at kv_tail = q0 - (q0 & 16).
// Output -> Ah in [B,S,H*DK] = [4096,1024] row-major for the final GEMM.
// ---------------------------------------------------------------------------
__global__ void __launch_bounds__(128)
attn_kernel(const _Float16* __restrict__ Qh,
            const _Float16* __restrict__ Kh,
            const _Float16* __restrict__ Vt,
            _Float16* __restrict__ Ah) {
    __shared__ __align__(16) _Float16 lds_p[4][16 * 32];

    const int lane = threadIdx.x & 31;
    const int wave = threadIdx.x >> 5;
    const int wid  = blockIdx.x * 4 + wave;   // 0 .. B*H*(S/16)-1
    const int qt = wid & 127;                 // query tile within (b,h)
    const int bh = wid >> 7;                  // 0..31
    const int q0 = qt * 16;

    const _Float16* Qp = Qh + (size_t)bh * S_ * DK_;
    const _Float16* Kp = Kh + (size_t)bh * S_ * DK_;
    const _Float16* Vp = Vt + (size_t)bh * DK_ * S_;

    const v16h aq0 = load_frag_a(Qp + (size_t)q0 * DK_ + 0,  DK_, lane);
    const v16h aq1 = load_frag_a(Qp + (size_t)q0 * DK_ + 32, DK_, lane);

    v8f o[4] = {};
    float mrun[8], lrun[8];
#pragma unroll
    for (int j = 0; j < 8; ++j) { mrun[j] = -1e30f; lrun[j] = 0.0f; }

    const int col = lane & 15, grp = lane >> 4;
    _Float16* pl = lds_p[wave];

    // Preload K fragments for block kv0 = 0.
    v16h kf[4];
    kf[0] = load_frag_b(Kp + 0,             DK_, lane);
    kf[1] = load_frag_b(Kp + 32,            DK_, lane);
    kf[2] = load_frag_b(Kp + 16 * DK_ + 0,  DK_, lane);
    kf[3] = load_frag_b(Kp + 16 * DK_ + 32, DK_, lane);

    const int kv_tail = q0 - (q0 & 16);   // the single masked diagonal block
    for (int kv0 = 0; kv0 < kv_tail; kv0 += 32) {
        attn_block<false>(Kp, Vp, pl, aq0, aq1, kf, kv0, kv0 + 32,
                          q0, lane, col, grp, o, mrun, lrun);
    }
    attn_block<true>(Kp, Vp, pl, aq0, aq1, kf, kv_tail, kv_tail,
                     q0, lane, col, grp, o, mrun, lrun);

    // ---- normalize and store: Ah[b*S + s][h*DK + d] ----
    const int b = bh >> 4, h = bh & 15;
#pragma unroll
    for (int c = 0; c < 4; ++c) {
        const int d = 16 * c + col;
#pragma unroll
        for (int j = 0; j < 8; ++j) {
            const int s = q0 + j + 8 * grp;
            Ah[((size_t)(b * S_ + s)) * D_ + h * DK_ + d] =
                (_Float16)(o[c][j] / lrun[j]);
        }
    }
}

// ---------------------------------------------------------------------------
// Kernel 3: output projection, f32 result.  out = Ah @ Wo^T + bo.
// ---------------------------------------------------------------------------
__global__ void __launch_bounds__(256)
out_proj_kernel(const _Float16* __restrict__ Ah,
                const _Float16* __restrict__ Woh,
                const float* __restrict__ bo,
                float* __restrict__ out) {
    const int lane = threadIdx.x & 31;
    const int wave = threadIdx.x >> 5;
    const int m0 = blockIdx.x * 128 + (wave & 3) * 32;
    const int n0 = blockIdx.y * 128 + (wave >> 2) * 64;

    v8f acc[2][4] = {};
    gemm_core_32x64(Ah + (size_t)m0 * D_, Woh + (size_t)n0 * D_, lane, acc);

    const int col = lane & 15, grp = lane >> 4;
#pragma unroll
    for (int r = 0; r < 2; ++r) {
#pragma unroll
        for (int c = 0; c < 4; ++c) {
            const int n = n0 + 16 * c + col;
            const float bn = bo[n];
#pragma unroll
            for (int j = 0; j < 8; ++j) {
                const int m = m0 + 16 * r + j + 8 * grp;
                out[(size_t)m * D_ + n] = acc[r][c][j] + bn;
            }
        }
    }
}

// ---------------------------------------------------------------------------
// Host-side launcher (graph-capture safe: only kernel launches on `stream`).
// Workspace layout (f16 elements, all 16B-aligned), total ~48 MB:
//   Xh[4096,1024] | W3[3][1024,1024] | Woh[1024,1024]
//   Qh,Kh [B,H,S,DK] | Vt [B,H,DK,S] | Ah [4096,1024]
// ---------------------------------------------------------------------------
extern "C" void kernel_launch(void* const* d_in, const int* in_sizes, int n_in,
                              void* d_out, int out_size, void* d_ws, size_t ws_size,
                              hipStream_t stream) {
    (void)in_sizes; (void)n_in; (void)out_size; (void)ws_size;

    const float* X  = (const float*)d_in[0];
    const float* Wq = (const float*)d_in[1];
    const float* bq = (const float*)d_in[2];
    const float* Wk = (const float*)d_in[3];
    const float* bk = (const float*)d_in[4];
    const float* Wv = (const float*)d_in[5];
    const float* bv = (const float*)d_in[6];
    const float* Wo = (const float*)d_in[7];
    const float* bo = (const float*)d_in[8];
    float* out = (float*)d_out;

    _Float16* ws = (_Float16*)d_ws;
    size_t off = 0;
    _Float16* Xh  = ws + off; off += (size_t)MTOT * D_;
    _Float16* W3  = ws + off; off += (size_t)3 * D_ * D_;
    _Float16* Woh = ws + off; off += (size_t)D_ * D_;
    _Float16* Qh  = ws + off; off += (size_t)B_ * H_ * S_ * DK_;
    _Float16* Kh  = ws + off; off += (size_t)B_ * H_ * S_ * DK_;
    _Float16* Vt  = ws + off; off += (size_t)B_ * H_ * S_ * DK_;
    _Float16* Ah  = ws + off; off += (size_t)MTOT * D_;

    // 0) converts
    cvt_f32_to_f16<<<2048, 256, 0, stream>>>(X,  Xh, MTOT * D_);
    cvt_f32_to_f16<<<512,  256, 0, stream>>>(Wq, W3, D_ * D_);
    cvt_f32_to_f16<<<512,  256, 0, stream>>>(Wk, W3 + (size_t)D_ * D_, D_ * D_);
    cvt_f32_to_f16<<<512,  256, 0, stream>>>(Wv, W3 + (size_t)2 * D_ * D_, D_ * D_);
    cvt_f32_to_f16<<<512,  256, 0, stream>>>(Wo, Woh, D_ * D_);

    // 1) fused QKV projection (z selects Q/K/V): blocks cover 128x128
    dim3 g1(MTOT / 128, D_ / 128, 3);
    qkv_proj_kernel<<<g1, 256, 0, stream>>>(Xh, W3, bq, bk, bv, Qh, Kh, Vt);

    // 2) causal flash attention: B*H*(S/16) = 4096 waves, 4 per block
    attn_kernel<<<(B_ * H_ * (S_ / 16)) / 4, 128, 0, stream>>>(Qh, Kh, Vt, Ah);

    // 3) output projection
    dim3 g2(MTOT / 128, D_ / 128);
    out_proj_kernel<<<g2, 256, 0, stream>>>(Ah, Woh, bo, out);
}
